// CausalTrajectoryPrediction_9620726743706
// MI455X (gfx1250) — compile-verified, matrix operand order checked by reading, not run on hardware
//
#include <hip/hip_runtime.h>

// CausalDerivative fused forward for MI455X (gfx1250, wave32, WMMA bf16).
// N=64 nodes, H=512, M=64, B=1024. One block = (node n, 64 batch rows).
// Pre-pass converts weights to bf16 in d_ws (if it fits) so the hot loops
// stream half the bytes from L2 and skip per-block f32->bf16 conversion.

#define NN 64
#define HH 512
#define MM 64
#define BT 64   // batch rows per block (4 row stripes of 16)

typedef __bf16 bf16_t;
typedef __attribute__((ext_vector_type(16))) __bf16 v16bf;
typedef __attribute__((ext_vector_type(8)))  __bf16 v8bf;
typedef __attribute__((ext_vector_type(8)))  float  v8f;

// A/B fragment (16x32 bf16, one lane's 16 halves). Lane's K values:
// {kb..kb+7} U {kb+16..kb+23} relative to k0, kb = 8*(lane>=16).
__device__ __forceinline__ v16bf fragT(const float* rowptr, int k0, int kb) {
  const float* p = rowptr + k0 + kb;
  v16bf f;
#pragma unroll
  for (int i = 0; i < 8; ++i) f[i] = (bf16_t)p[i];
#pragma unroll
  for (int i = 0; i < 8; ++i) f[i + 8] = (bf16_t)p[16 + i];
  return f;
}
__device__ __forceinline__ v16bf fragT(const bf16_t* rowptr, int k0, int kb) {
  const bf16_t* p = rowptr + k0 + kb;
  v8bf lo = *(const v8bf*)(p);
  v8bf hi = *(const v8bf*)(p + 16);
  v16bf f;
#pragma unroll
  for (int i = 0; i < 8; ++i) { f[i] = lo[i]; f[i + 8] = hi[i]; }
  return f;
}

#define WMMA_BF16(A, B, C) \
  __builtin_amdgcn_wmma_f32_16x16x32_bf16(false, (A), false, (B), (short)0, (C), false, false)

// ---- pre-pass: convert W1, W2, W3[:, :, :64] to bf16 in workspace ----
__global__ __launch_bounds__(256) void convert_weights(
    const float* __restrict__ W1, const float* __restrict__ W2,
    const float* __restrict__ W3,
    bf16_t* __restrict__ W1b, bf16_t* __restrict__ W2b, bf16_t* __restrict__ W3ab) {
  long i = (long)blockIdx.x * 256 + threadIdx.x;   // 0 .. 64*512*64-1
  if (i < (long)NN * HH * NN) {
    W1b[i] = (bf16_t)W1[i];
    W2b[i] = (bf16_t)W2[i];
    long row = i >> 6, k = i & 63;                 // W3 rows have ld = 128
    W3ab[i] = (bf16_t)W3[row * (MM + NN) + k];
  }
}

template <typename WT>
__global__ __launch_bounds__(256) void causal_deriv_kernel(
    const float* __restrict__ x,
    const WT* __restrict__ W1, const WT* __restrict__ W2,
    const WT* __restrict__ W3a, int ld3,
    const float* __restrict__ W3full, const float* __restrict__ b3,
    const float* __restrict__ W4, const float* __restrict__ b4,
    float* __restrict__ out) {
  __shared__ bf16_t sA[BT * NN];   // x tile (col n zeroed), bf16
  __shared__ bf16_t sH[BT * HH];   // h1 tile (relu'd), bf16  (64 KB)
  __shared__ bf16_t sR[BT * MM];   // r1 tile (relu'd), bf16
  __shared__ float  sXn[BT];       // x[:, n]
  __shared__ float  sOut[BT];      // final dot accumulators

  const int n    = blockIdx.y;
  const int b0   = blockIdx.x * BT;
  const int tid  = threadIdx.x;
  const int wv   = tid >> 5;
  const int lane = tid & 31;
  const int l15  = lane & 15;
  const int hi8  = (lane & 16) ? 8 : 0;   // K-base for frags AND row offset for C/D
  const int kb   = hi8;

  // ---- stage 0: stage x tile into LDS as bf16, zero own column ----
  for (int idx = tid; idx < BT * NN; idx += 256) {
    int r = idx >> 6, c = idx & 63;
    float v = x[(b0 + r) * NN + c];
    if (c == n) { sXn[r] = v; sA[idx] = (bf16_t)0.0f; }
    else        { sA[idx] = (bf16_t)v; }
  }
  if (tid < BT) sOut[tid] = 0.0f;
  __syncthreads();

  // ---- stage 1: h1 = relu(x~ . W1[n]^T)  (64x64)x(64x512) ----
  // Wave w owns cols [64w, 64w+64): each weight element loaded exactly once per block.
  {
    const int colb = wv * 64;
    v8f acc[4][4];                        // [row stripe][col tile]
#pragma unroll
    for (int s = 0; s < 4; ++s)
#pragma unroll
      for (int t = 0; t < 4; ++t)
#pragma unroll
        for (int j = 0; j < 8; ++j) acc[s][t][j] = 0.0f;

#pragma unroll
    for (int k0 = 0; k0 < NN; k0 += 32) {
      v16bf a[4];
#pragma unroll
      for (int s = 0; s < 4; ++s) a[s] = fragT(sA + (s * 16 + l15) * NN, k0, kb);
#pragma unroll
      for (int t = 0; t < 4; ++t) {
        int col = colb + t * 16 + l15;
        v16bf bfr = fragT(W1 + ((long)n * HH + col) * NN, k0, kb);
#pragma unroll
        for (int s = 0; s < 4; ++s) acc[s][t] = WMMA_BF16(a[s], bfr, acc[s][t]);
      }
    }
#pragma unroll
    for (int s = 0; s < 4; ++s)
#pragma unroll
      for (int t = 0; t < 4; ++t) {
        int col = colb + t * 16 + l15;
#pragma unroll
        for (int j = 0; j < 8; ++j) {
          float v = acc[s][t][j];
          sH[(s * 16 + j + hi8) * HH + col] = (bf16_t)(v > 0.0f ? v : 0.0f);
        }
      }
  }
  __syncthreads();

  // ---- stage 2: r1 = relu(h1 . W2[n]^T)  (64x512)x(512x64) ----
  {
    const int ct = wv & 3;                // col tile 0..3
    const int sp = wv >> 2;               // stripes sp, sp+2
    v8f acc2[2];
#pragma unroll
    for (int q = 0; q < 2; ++q)
#pragma unroll
      for (int j = 0; j < 8; ++j) acc2[q][j] = 0.0f;

    const WT* w2row = W2 + ((long)n * MM + (ct * 16 + l15)) * HH;
#pragma unroll
    for (int k0 = 0; k0 < HH; k0 += 32) {
      v16bf bfr = fragT(w2row, k0, kb);
#pragma unroll
      for (int q = 0; q < 2; ++q) {
        int s = sp + 2 * q;
        v16bf a = fragT(sH + (s * 16 + l15) * HH, k0, kb);
        acc2[q] = WMMA_BF16(a, bfr, acc2[q]);
      }
    }
#pragma unroll
    for (int q = 0; q < 2; ++q) {
      int s = sp + 2 * q;
#pragma unroll
      for (int j = 0; j < 8; ++j) {
        float v = acc2[q][j];
        sR[(s * 16 + j + hi8) * MM + (ct * 16 + l15)] = (bf16_t)(v > 0.0f ? v : 0.0f);
      }
    }
  }
  __syncthreads();

  // ---- stage 3+4: h3 = relu(r1.W3a^T + x_n (x) w3c + b3); out = relu(h3.W4 + b4) ----
  {
    const int colb = wv * 64;
    float bias[4], w3c[4], w4c[4];
#pragma unroll
    for (int t = 0; t < 4; ++t) {
      int col = colb + t * 16 + l15;
      bias[t] = b3[n * HH + col];
      w3c[t]  = W3full[((long)n * HH + col) * (MM + NN) + MM + n];
      w4c[t]  = W4[n * HH + col];
    }
    v8f acc[4][4];
#pragma unroll
    for (int s = 0; s < 4; ++s)
#pragma unroll
      for (int j = 0; j < 8; ++j) {
        float xr = sXn[s * 16 + j + hi8];
#pragma unroll
        for (int t = 0; t < 4; ++t) acc[s][t][j] = bias[t] + xr * w3c[t];
      }

#pragma unroll
    for (int k0 = 0; k0 < MM; k0 += 32) {
      v16bf a[4];
#pragma unroll
      for (int s = 0; s < 4; ++s) a[s] = fragT(sR + (s * 16 + l15) * MM, k0, kb);
#pragma unroll
      for (int t = 0; t < 4; ++t) {
        int col = colb + t * 16 + l15;
        v16bf bfr = fragT(W3a + ((long)n * HH + col) * ld3, k0, kb);
#pragma unroll
        for (int s = 0; s < 4; ++s) acc[s][t] = WMMA_BF16(a[s], bfr, acc[s][t]);
      }
    }

    // fused GEMV vs W4, one row stripe at a time to limit live registers
#pragma unroll
    for (int s = 0; s < 4; ++s) {
      float p[8];
#pragma unroll
      for (int j = 0; j < 8; ++j) p[j] = 0.0f;
#pragma unroll
      for (int t = 0; t < 4; ++t)
#pragma unroll
        for (int j = 0; j < 8; ++j) {
          float v = acc[s][t][j];
          p[j] += (v > 0.0f ? v : 0.0f) * w4c[t];
        }
#pragma unroll
      for (int m = 8; m >= 1; m >>= 1)
#pragma unroll
        for (int j = 0; j < 8; ++j) p[j] += __shfl_xor(p[j], m, 32);
      if (l15 == 0) {
#pragma unroll
        for (int j = 0; j < 8; ++j) atomicAdd(&sOut[s * 16 + j + hi8], p[j]);
      }
    }
  }
  __syncthreads();

  if (tid < BT) {
    float v = sOut[tid] + b4[n];
    out[(b0 + tid) * NN + n] = v > 0.0f ? v : 0.0f;
  }
}

extern "C" void kernel_launch(void* const* d_in, const int* in_sizes, int n_in,
                              void* d_out, int out_size, void* d_ws, size_t ws_size,
                              hipStream_t stream) {
  const float* x  = (const float*)d_in[0];
  const float* W1 = (const float*)d_in[1];
  const float* W2 = (const float*)d_in[2];
  const float* W3 = (const float*)d_in[3];
  const float* b3 = (const float*)d_in[4];
  const float* W4 = (const float*)d_in[5];
  const float* b4 = (const float*)d_in[6];
  float* out = (float*)d_out;
  (void)in_sizes; (void)n_in; (void)out_size;

  dim3 grid(1024 / BT, NN);   // (16 batch tiles, 64 nodes)
  const long NE = (long)NN * HH * NN;              // 2,097,152 per weight array
  if (ws_size >= (size_t)(3 * NE * sizeof(bf16_t))) {
    bf16_t* W1b  = (bf16_t*)d_ws;
    bf16_t* W2b  = W1b + NE;
    bf16_t* W3ab = W2b + NE;
    convert_weights<<<(int)((NE + 255) / 256), 256, 0, stream>>>(W1, W2, W3, W1b, W2b, W3ab);
    causal_deriv_kernel<bf16_t><<<grid, 256, 0, stream>>>(
        x, W1b, W2b, W3ab, NN, W3, b3, W4, b4, out);
  } else {
    causal_deriv_kernel<float><<<grid, 256, 0, stream>>>(
        x, W1, W2, W3, MM + NN, W3, b3, W4, b4, out);
  }
}
